// SelfAttention_33200097198147
// MI455X (gfx1250) — matrix operand order repaired and checked by reading, not compiled, not run on hardware
//
#include <hip/hip_runtime.h>
#include <hip/hip_bf16.h>

typedef __bf16 bf16;
typedef bf16  v16bf __attribute__((ext_vector_type(16)));
typedef bf16  v8bf  __attribute__((ext_vector_type(8)));
typedef float v8f   __attribute__((ext_vector_type(8)));

#define B_  2
#define N_  2048
#define D_  768
#define H_  12
#define DH_ 64
#define ROWS_ (B_ * N_)          // 4096

__device__ __forceinline__ v8f wmma_bf16(v16bf a, v16bf b, v8f c) {
  return __builtin_amdgcn_wmma_f32_16x16x32_bf16(false, a, false, b, (short)0, c,
                                                 false, false);
}

// A-fragment: 16x32 bf16 tile at (row0, k0), row-major src with leading dim ld.
// Lane r=lane&15 holds row r; K = k0 + hi*8 + {0..7} and k0 + 16 + hi*8 + {0..7}.
__device__ __forceinline__ v16bf load_fragA(const bf16* __restrict__ base,
                                            int row0, int k0, int ld, int lane) {
  int r = lane & 15, hi = lane >> 4;
  const bf16* p = base + (size_t)(row0 + r) * ld + k0 + hi * 8;
  union { v16bf v; v8bf h[2]; } u;
  u.h[0] = *(const v8bf*)(p);
  u.h[1] = *(const v8bf*)(p + 16);
  return u.v;
}

// B-fragment: 32x16 bf16. Column n = n0 + (lane&15) is a row of the row-major
// source (stride ld, K contiguous); K = k0 + hi*8 + {0..7} and k0 + 16 + hi*8 + {0..7}.
__device__ __forceinline__ v16bf load_fragB(const bf16* __restrict__ base,
                                            int n0, int k0, int ld, int lane) {
  int c = lane & 15, hi = lane >> 4;
  const bf16* p = base + (size_t)(n0 + c) * ld + k0 + hi * 8;
  union { v16bf v; v8bf h[2]; } u;
  u.h[0] = *(const v8bf*)(p);
  u.h[1] = *(const v8bf*)(p + 16);
  return u.v;
}

// ---------------------------------------------------------------- fp32 -> bf16
__global__ void cvt_bf16_kernel(const float* __restrict__ in,
                                bf16* __restrict__ out, int n8) {
  int i = blockIdx.x * blockDim.x + threadIdx.x;
  if (i >= n8) return;
  const float4* p = (const float4*)in + (size_t)i * 2;
  float4 a = p[0], b = p[1];
  v8bf r;
  r[0] = (bf16)a.x; r[1] = (bf16)a.y; r[2] = (bf16)a.z; r[3] = (bf16)a.w;
  r[4] = (bf16)b.x; r[5] = (bf16)b.y; r[6] = (bf16)b.z; r[7] = (bf16)b.w;
  *(v8bf*)(out + (size_t)i * 8) = r;
}

// ------------------------------------------- QKV GEMM + RoPE + head reshaping
// One wave computes a 32x64 tile of qkv = x @ qkv_w^T (8 WMMAs / 12 loads per
// k-step; B-fragments reused across both 16-row halves). 64 cols == one
// (sec,head). Q/K: RoPE applied in-register (d <-> d+32 are acc j <-> j^2),
// written to Qb/Kb [B,H,N,64] bf16. V written transposed to Vt [B,H,64,N].
__global__ __launch_bounds__(128) void qkv_rope_kernel(
    const bf16* __restrict__ xb, const bf16* __restrict__ wb,
    const float* __restrict__ sin_t, const float* __restrict__ cos_t,
    bf16* __restrict__ Qb, bf16* __restrict__ Kb, bf16* __restrict__ Vt) {
  int lane = threadIdx.x & 31;
  int wv = threadIdx.x >> 5;
  int t = blockIdx.x * 4 + wv;            // 128 row-tiles * 36 col-tiles = 4608
  int coltile = t % 36;
  int rowtile = t / 36;
  int mbase = rowtile * 32;
  int ebase = coltile * 64;

  v8f acc[8] = {};                        // [rowhalf*4 + j]
#pragma unroll 1
  for (int k0 = 0; k0 < D_; k0 += 32) {
    v16bf a0 = load_fragA(xb, mbase,      k0, D_, lane);
    v16bf a1 = load_fragA(xb, mbase + 16, k0, D_, lane);
#pragma unroll
    for (int j = 0; j < 4; ++j) {
      v16bf bf = load_fragB(wb, ebase + j * 16, k0, D_, lane);
      acc[j]     = wmma_bf16(a0, bf, acc[j]);
      acc[4 + j] = wmma_bf16(a1, bf, acc[4 + j]);
    }
  }

  int c = lane & 15, hif = lane >> 4;
  int b = mbase >> 11;                    // / N_
  int nseq0 = mbase & (N_ - 1);
  int sec = ebase / D_;                   // 0=Q 1=K 2=V
  int h = (ebase % D_) / DH_;

  if (sec < 2) {
    bf16* dst = (sec == 0) ? Qb : Kb;
    dst += ((size_t)(b * H_ + h)) * N_ * DH_;
#pragma unroll
    for (int rh = 0; rh < 2; ++rh) {
#pragma unroll
      for (int j = 0; j < 4; ++j) {
#pragma unroll
        for (int i = 0; i < 8; ++i) {
          int n = nseq0 + rh * 16 + i + 8 * hif;
          int d = j * 16 + c;
          float v = acc[rh * 4 + j][i];
          float rot = (j < 2) ? -acc[rh * 4 + j + 2][i] : acc[rh * 4 + j - 2][i];
          float sv = sin_t[n * DH_ + d];
          float cv = cos_t[n * DH_ + d];
          dst[(size_t)n * DH_ + d] = (bf16)(v * cv + rot * sv);
        }
      }
    }
  } else {
    bf16* dst = Vt + ((size_t)(b * H_ + h)) * DH_ * N_;
#pragma unroll
    for (int rh = 0; rh < 2; ++rh) {
#pragma unroll
      for (int j = 0; j < 4; ++j) {
#pragma unroll
        for (int i = 0; i < 8; ++i) {
          int n = nseq0 + rh * 16 + i + 8 * hif;
          int d = j * 16 + c;
          dst[(size_t)d * N_ + n] = (bf16)acc[rh * 4 + j][i];
        }
      }
    }
  }
}

// ---------------------------------------------------- flash attention (bf16)
// One wave owns 16 query rows; streams 32-key blocks with online softmax.
// 4 waves/block share the same (b,h) so K/V tiles hit the WGP cache.
__global__ __launch_bounds__(128) void attn_kernel(
    const bf16* __restrict__ Qb, const bf16* __restrict__ Kb,
    const bf16* __restrict__ Vt, bf16* __restrict__ attnb) {
  __shared__ alignas(16) bf16 plds[4][16 * 32];
  int lane = threadIdx.x & 31;
  int wv = threadIdx.x >> 5;
  int t = blockIdx.x * 4 + wv;            // 24 * 128 = 3072 q-tiles
  int bh = t >> 7;                        // 128 q-tiles per (b,h)
  int m0 = (t & 127) * 16;

  const bf16* Qh = Qb + (size_t)bh * N_ * DH_;
  const bf16* Kh = Kb + (size_t)bh * N_ * DH_;
  const bf16* Vh = Vt + (size_t)bh * DH_ * N_;

  v16bf qa0 = load_fragA(Qh, m0, 0, DH_, lane);
  v16bf qa1 = load_fragA(Qh, m0, 32, DH_, lane);

  v8f o[4] = {};
  float mrow[8], lrow[8];
#pragma unroll
  for (int i = 0; i < 8; ++i) { mrow[i] = -1e30f; lrow[i] = 0.0f; }
  const float scale = 0.125f;             // Dh^-0.5
  int c = lane & 15, hif = lane >> 4;

#pragma unroll 1
  for (int kb = 0; kb < N_; kb += 32) {
    v16bf kf00 = load_fragB(Kh, kb,      0,  DH_, lane);
    v16bf kf01 = load_fragB(Kh, kb,      32, DH_, lane);
    v16bf kf10 = load_fragB(Kh, kb + 16, 0,  DH_, lane);
    v16bf kf11 = load_fragB(Kh, kb + 16, 32, DH_, lane);
    v8f s0 = {}, s1 = {};
    s0 = wmma_bf16(qa0, kf00, s0);
    s0 = wmma_bf16(qa1, kf01, s0);
    s1 = wmma_bf16(qa0, kf10, s1);
    s1 = wmma_bf16(qa1, kf11, s1);

    float p0[8], p1[8];
#pragma unroll
    for (int i = 0; i < 8; ++i) {
      float a = s0[i] * scale, bv = s1[i] * scale;
      float mx = fmaxf(a, bv);
      mx = fmaxf(mx, __shfl_xor(mx, 1, 32));
      mx = fmaxf(mx, __shfl_xor(mx, 2, 32));
      mx = fmaxf(mx, __shfl_xor(mx, 4, 32));
      mx = fmaxf(mx, __shfl_xor(mx, 8, 32));
      float mnew = fmaxf(mrow[i], mx);
      float corr = __expf(mrow[i] - mnew);
      float e0 = __expf(a - mnew), e1 = __expf(bv - mnew);
      float rs = e0 + e1;
      rs += __shfl_xor(rs, 1, 32);
      rs += __shfl_xor(rs, 2, 32);
      rs += __shfl_xor(rs, 4, 32);
      rs += __shfl_xor(rs, 8, 32);
      lrow[i] = lrow[i] * corr + rs;
      mrow[i] = mnew;
      p0[i] = e0; p1[i] = e1;
      o[0][i] *= corr; o[1][i] *= corr; o[2][i] *= corr; o[3][i] *= corr;
    }

    // C-layout P -> LDS (row-major 16x32 bf16) -> A-fragment layout
#pragma unroll
    for (int i = 0; i < 8; ++i) {
      int m = i + 8 * hif;
      plds[wv][m * 32 + c]      = (bf16)p0[i];
      plds[wv][m * 32 + 16 + c] = (bf16)p1[i];
    }
    __syncthreads();
    v16bf pa;
    {
      int r = lane & 15;
      const bf16* pr = &plds[wv][r * 32 + hif * 8];
      union { v16bf v; v8bf h[2]; } u;
      u.h[0] = *(const v8bf*)(pr);
      u.h[1] = *(const v8bf*)(pr + 16);
      pa = u.v;
    }
    __syncthreads();

#pragma unroll
    for (int j = 0; j < 4; ++j) {
      v16bf vb = load_fragB(Vh, j * 16, kb, N_, lane);  // Vt: K (=key) contiguous
      o[j] = wmma_bf16(pa, vb, o[j]);
    }
  }

  int h = bh % H_, b = bh / H_;
#pragma unroll
  for (int j = 0; j < 4; ++j) {
#pragma unroll
    for (int i = 0; i < 8; ++i) {
      int m = i + 8 * hif;
      int n = m0 + m;
      float val = o[j][i] / lrow[i];
      attnb[((size_t)(b * N_ + n)) * D_ + h * DH_ + j * 16 + c] = (bf16)val;
    }
  }
}

// ------------------------------------------------------ out projection + bias
// 32x64 tile per wave (8 WMMAs / 12 loads per k-step).
__global__ __launch_bounds__(128) void proj_kernel(
    const bf16* __restrict__ ab, const bf16* __restrict__ wb,
    const float* __restrict__ bias, float* __restrict__ out) {
  int lane = threadIdx.x & 31;
  int wv = threadIdx.x >> 5;
  int t = blockIdx.x * 4 + wv;            // 128 * 12 = 1536
  int coltile = t % 12;
  int rowtile = t / 12;
  int mbase = rowtile * 32, nbase = coltile * 64;

  v8f acc[8] = {};
#pragma unroll 1
  for (int k0 = 0; k0 < D_; k0 += 32) {
    v16bf a0 = load_fragA(ab, mbase,      k0, D_, lane);
    v16bf a1 = load_fragA(ab, mbase + 16, k0, D_, lane);
#pragma unroll
    for (int j = 0; j < 4; ++j) {
      v16bf bf = load_fragB(wb, nbase + j * 16, k0, D_, lane);
      acc[j]     = wmma_bf16(a0, bf, acc[j]);
      acc[4 + j] = wmma_bf16(a1, bf, acc[4 + j]);
    }
  }
  int c = lane & 15, hif = lane >> 4;
#pragma unroll
  for (int rh = 0; rh < 2; ++rh) {
#pragma unroll
    for (int j = 0; j < 4; ++j) {
#pragma unroll
      for (int i = 0; i < 8; ++i) {
        int m = mbase + rh * 16 + i + 8 * hif;
        int n = nbase + j * 16 + c;
        out[(size_t)m * D_ + n] = acc[rh * 4 + j][i] + bias[n];
      }
    }
  }
}

extern "C" void kernel_launch(void* const* d_in, const int* in_sizes, int n_in,
                              void* d_out, int out_size, void* d_ws, size_t ws_size,
                              hipStream_t stream) {
  const float* x      = (const float*)d_in[0];
  const float* sin_t  = (const float*)d_in[1];
  const float* cos_t  = (const float*)d_in[2];
  const float* qkv_w  = (const float*)d_in[3];
  const float* proj_w = (const float*)d_in[4];
  const float* proj_b = (const float*)d_in[5];
  float* out = (float*)d_out;

  char* ws = (char*)d_ws;
  size_t off = 0;
  bf16* xb    = (bf16*)(ws + off); off += (size_t)ROWS_ * D_ * 2;        // 6.29 MB
  bf16* wqkvb = (bf16*)(ws + off); off += (size_t)3 * D_ * D_ * 2;       // 3.54 MB
  bf16* wprjb = (bf16*)(ws + off); off += (size_t)D_ * D_ * 2;           // 1.18 MB
  bf16* Qb    = (bf16*)(ws + off); off += (size_t)B_ * H_ * N_ * DH_ * 2;
  bf16* Kb    = (bf16*)(ws + off); off += (size_t)B_ * H_ * N_ * DH_ * 2;
  bf16* Vt    = (bf16*)(ws + off); off += (size_t)B_ * H_ * DH_ * N_ * 2;
  bf16* attnb = (bf16*)(ws + off); off += (size_t)ROWS_ * D_ * 2;        // total ~36.2 MB

  int n8x = ROWS_ * D_ / 8, n8q = 3 * D_ * D_ / 8, n8p = D_ * D_ / 8;
  cvt_bf16_kernel<<<(n8x + 255) / 256, 256, 0, stream>>>(x, xb, n8x);
  cvt_bf16_kernel<<<(n8q + 255) / 256, 256, 0, stream>>>(qkv_w, wqkvb, n8q);
  cvt_bf16_kernel<<<(n8p + 255) / 256, 256, 0, stream>>>(proj_w, wprjb, n8p);

  qkv_rope_kernel<<<(128 * 36) / 4, 128, 0, stream>>>(xb, wqkvb, sin_t, cos_t,
                                                      Qb, Kb, Vt);
  attn_kernel<<<(B_ * H_ * (N_ / 16)) / 4, 128, 0, stream>>>(Qb, Kb, Vt, attnb);
  proj_kernel<<<(128 * 12) / 4, 128, 0, stream>>>(attnb, wprjb, proj_b, out);
}
